// EdgeDegreeEmbedding_86629490360995
// MI455X (gfx1250) — matrix operand order repaired and checked by reading, not compiled, not run on hardware
//
#include <hip/hip_runtime.h>

// ---------------- CDNA5 WMMA types ----------------
typedef __attribute__((ext_vector_type(16))) __bf16 v16bf;
typedef __attribute__((ext_vector_type(8)))  __bf16 v8bf;
typedef __attribute__((ext_vector_type(8)))  float  v8f;

__device__ __forceinline__ __bf16 f2bf(float f) { return (__bf16)f; }

// ---------------- problem constants ----------------
#define N_NODES 1600
#define N_EDGES 32000
#define CCH     128
#define M0      7
#define N_SPH   49
#define N_ALL   29
#define BASIS   512
#define HID     128
#define D_IN    768
#define W3N     896
#define MT      16           // edges per block
#define INV_RESCALE (1.0f / 23.395238876342773f)

// workspace (bf16 weight packs in WMMA-B lane layout)
#define WS_W1   0            // 768*128 bf16 = 196608 B
#define WS_W2   196608       // 128*128 bf16 =  32768 B
#define WS_W3   229376       // 128*896 bf16 = 229376 B
// total ws use: 458752 B

// LDS layout (temporal aliasing), total 57664 B
#define OFF_XFEP  0        // [16 rows][24 kt][2 hi][16] bf16 = 24576 (stage-1 A, packed)
#define OFF_WIG   0        // [16 e][64 gi][8] bf16 = 16384 (aliases dead XFEP)
#define OFF_HBP   16384    // [16 rows][4 kt][2 hi][16] bf16 = 4096
#define OFF_HC    24576    // [16][128] f32 = 8192
#define OFF_X0T   24576    // [16 e][128 ch][8 m] bf16 = 32768 (aliases dead HC)
#define OFF_TGT   57344
#define OFF_MU    57408
#define OFF_RS    57472
#define OFF_SRCZ  57536
#define OFF_TGTZ  57600
#define SMEM_SZ   57664

// zero-init output (harness poisons d_out; we accumulate with atomics)
__global__ void zero_out_kernel(float4* __restrict__ out, int n4) {
    int i = blockIdx.x * blockDim.x + threadIdx.x;
    if (i < n4) out[i] = make_float4(0.f, 0.f, 0.f, 0.f);
}

// Pack a row-major f32 [K][N] weight into bf16 WMMA-B fragments:
//   dst[((nt*NKT + kt)*32 + lane)*16 + i] = src[(kt*32 + i + (lane>>4)*16)*N + nt*16 + (lane&15)]
// so each lane's 16 B elements are 32 contiguous bytes (two b128 loads).
__global__ void pack_w_kernel(const float* __restrict__ src, __bf16* __restrict__ dst,
                              int K, int N, int total) {
    int o = blockIdx.x * blockDim.x + threadIdx.x;
    if (o >= total) return;
    int i    = o & 15;
    int lane = (o >> 4) & 31;
    int t    = o >> 9;
    int NKT  = K >> 5;
    int kt   = t % NKT;
    int nt   = t / NKT;
    int k = kt * 32 + i + ((lane >> 4) & 1) * 16;
    int n = nt * 16 + (lane & 15);
    dst[o] = (__bf16)src[k * N + n];
}

// inverse of the A-fragment k-map: k (0..31) -> (hi, elem)
//   hi = (k>>4)&1 ; elem = k&15
__device__ __forceinline__ int a_inv_hi(int k)   { return (k >> 4) & 1; }
__device__ __forceinline__ int a_inv_elem(int k) { return k & 15; }

__global__ __launch_bounds__(256)
void edge_degree_fused_kernel(const int*   __restrict__ atomic_numbers,
                              const float* __restrict__ edge_distance,
                              const int*   __restrict__ edge_index,
                              const float* __restrict__ src_table,
                              const float* __restrict__ tgt_table,
                              const __bf16* __restrict__ w1p,
                              const __bf16* __restrict__ w2p,
                              const __bf16* __restrict__ w3p,
                              const float* __restrict__ b1,
                              const float* __restrict__ g1, const float* __restrict__ be1,
                              const float* __restrict__ b2,
                              const float* __restrict__ g2, const float* __restrict__ be2,
                              const float* __restrict__ b3,
                              const float* __restrict__ wigner_inv,
                              float*       __restrict__ out) {
    __shared__ __align__(32) unsigned char smem[SMEM_SZ];
    __bf16* xfeP  = (__bf16*)(smem + OFF_XFEP);
    __bf16* wig   = (__bf16*)(smem + OFF_WIG);
    __bf16* hbP   = (__bf16*)(smem + OFF_HBP);
    float*  hC    = (float*) (smem + OFF_HC);
    __bf16* x0t   = (__bf16*)(smem + OFF_X0T);
    int*    tgtid = (int*)   (smem + OFF_TGT);
    float*  muv   = (float*) (smem + OFF_MU);
    float*  rsv   = (float*) (smem + OFF_RS);
    int*    srcz  = (int*)   (smem + OFF_SRCZ);
    int*    tgtz  = (int*)   (smem + OFF_TGTZ);

    const int tid  = threadIdx.x;
    const int lane = tid & 31;
    const int wv   = tid >> 5;
    const int hi   = (lane >> 4) & 1;
    const int lr   = lane & 15;
    const int e0   = blockIdx.x * MT;

    // -------- epoch 1: edge metadata --------
    if (tid < MT) {
        int e = e0 + tid;
        int s = edge_index[e];
        int t = edge_index[N_EDGES + e];
        srcz[tid]  = atomic_numbers[s];
        tgtz[tid]  = atomic_numbers[t];
        tgtid[tid] = t;
    }
    __syncthreads();

    // -------- epoch 2: features -> LDS, pre-permuted into A fragments --------
    for (int idx = tid; idx < MT * D_IN; idx += 256) {
        int r = idx / D_IN, c = idx - r * D_IN;
        float v;
        if (c < BASIS)            v = edge_distance[(e0 + r) * BASIS + c];
        else if (c < BASIS + HID) v = src_table[srcz[r] * HID + (c - BASIS)];
        else                      v = tgt_table[tgtz[r] * HID + (c - BASIS - HID)];
        int kt = c >> 5, cc = c & 31;
        xfeP[((r * 24 + kt) * 2 + a_inv_hi(cc)) * 16 + a_inv_elem(cc)] = f2bf(v);
    }
    __syncthreads();

    // -------- epoch 3: GEMM1 [16,768]x[768,128] (wave -> 16 cols) --------
    {
        v8f acc = {0.f,0.f,0.f,0.f,0.f,0.f,0.f,0.f};
        #pragma unroll 4
        for (int kt = 0; kt < 24; ++kt) {
            v16bf a = *(const v16bf*)(xfeP + ((lr * 24 + kt) * 2 + hi) * 16);
            v16bf b = *(const v16bf*)(w1p  + ((wv * 24 + kt) * 32 + lane) * 16);
            acc = __builtin_amdgcn_wmma_f32_16x16x32_bf16(false, a, false, b,
                                                          (short)0, acc, false, false);
        }
        const int col = wv * 16 + lr;
        const float bias = b1[col];
        #pragma unroll
        for (int r = 0; r < 8; ++r) hC[(r + hi * 8) * HID + col] = acc[r] + bias;
    }
    __syncthreads();

    // -------- epoch 4: LN1 stats (tid<16)  +  Wigner columns -> LDS (all) --------
    if (tid < MT) {
        float s = 0.f, s2 = 0.f;
        for (int c = 0; c < HID; ++c) { float v = hC[tid * HID + c]; s += v; s2 += v * v; }
        float m = s * (1.f / HID);
        muv[tid] = m;
        rsv[tid] = rsqrtf(s2 * (1.f / HID) - m * m + 1e-5f);
    }
    // wig[e][gi][m] padded to [16][64][8], zero outside (gi<49, m<7)
    for (int idx = tid; idx < MT * 64 * 8; idx += 256) {
        int e = idx >> 9, rem = idx & 511;
        int gi = rem >> 3, m = rem & 7;
        float v = 0.f;
        if (gi < N_SPH && m < M0) {
            int colc = (m <= 2) ? (m * m + m) : (5 * m - 4);
            v = wigner_inv[((e0 + e) * N_SPH + gi) * N_ALL + colc];
        }
        wig[idx] = f2bf(v);
    }
    __syncthreads();

    // -------- epoch 5: LN1 apply + SiLU -> hbP (A-fragment layout) --------
    for (int idx = tid; idx < MT * HID; idx += 256) {
        int r = idx >> 7, c = idx & 127;
        float v = (hC[idx] - muv[r]) * rsv[r] * g1[c] + be1[c];
        v = v / (1.f + __expf(-v));
        int kt = c >> 5, cc = c & 31;
        hbP[((r * 4 + kt) * 2 + a_inv_hi(cc)) * 16 + a_inv_elem(cc)] = f2bf(v);
    }
    __syncthreads();

    // -------- epoch 6: GEMM2 [16,128]x[128,128] --------
    {
        v8f acc = {0.f,0.f,0.f,0.f,0.f,0.f,0.f,0.f};
        #pragma unroll
        for (int kt = 0; kt < 4; ++kt) {
            v16bf a = *(const v16bf*)(hbP + ((lr * 4 + kt) * 2 + hi) * 16);
            v16bf b = *(const v16bf*)(w2p + ((wv * 4 + kt) * 32 + lane) * 16);
            acc = __builtin_amdgcn_wmma_f32_16x16x32_bf16(false, a, false, b,
                                                          (short)0, acc, false, false);
        }
        const int col = wv * 16 + lr;
        const float bias = b2[col];
        #pragma unroll
        for (int r = 0; r < 8; ++r) hC[(r + hi * 8) * HID + col] = acc[r] + bias;
    }
    __syncthreads();
    // -------- epoch 7: LN2 stats --------
    if (tid < MT) {
        float s = 0.f, s2 = 0.f;
        for (int c = 0; c < HID; ++c) { float v = hC[tid * HID + c]; s += v; s2 += v * v; }
        float m = s * (1.f / HID);
        muv[tid] = m;
        rsv[tid] = rsqrtf(s2 * (1.f / HID) - m * m + 1e-5f);
    }
    __syncthreads();
    // -------- epoch 8: LN2 apply + SiLU -> hbP --------
    for (int idx = tid; idx < MT * HID; idx += 256) {
        int r = idx >> 7, c = idx & 127;
        float v = (hC[idx] - muv[r]) * rsv[r] * g2[c] + be2[c];
        v = v / (1.f + __expf(-v));
        int kt = c >> 5, cc = c & 31;
        hbP[((r * 4 + kt) * 2 + a_inv_hi(cc)) * 16 + a_inv_elem(cc)] = f2bf(v);
    }
    __syncthreads();

    // -------- epoch 9: GEMM3 [16,128]x[128,896] -> x0t[e][ch][m] (pad m=7 zeroed) --------
    for (int idx = tid; idx < MT * CCH; idx += 256)   // zero the m=7 pad lane
        x0t[idx * 8 + 7] = f2bf(0.f);
    {
        v16bf a3[4];
        #pragma unroll
        for (int kt = 0; kt < 4; ++kt)
            a3[kt] = *(const v16bf*)(hbP + ((lr * 4 + kt) * 2 + hi) * 16);
        for (int nt = wv; nt < 56; nt += 8) {
            v8f acc = {0.f,0.f,0.f,0.f,0.f,0.f,0.f,0.f};
            #pragma unroll
            for (int kt = 0; kt < 4; ++kt) {
                v16bf b = *(const v16bf*)(w3p + ((nt * 4 + kt) * 32 + lane) * 16);
                acc = __builtin_amdgcn_wmma_f32_16x16x32_bf16(false, a3[kt], false, b,
                                                              (short)0, acc, false, false);
            }
            const int col = nt * 16 + lr;      // col = m*128 + ch
            const int mm = col >> 7, ch = col & 127;
            const float bias = b3[col];
            #pragma unroll
            for (int r = 0; r < 8; ++r) {
                int edge = r + hi * 8;
                x0t[(edge * CCH + ch) * 8 + mm] = f2bf(acc[r] + bias);
            }
        }
    }
    __syncthreads();

    // -------- epoch 10: per-edge Wigner [49x7]@[7x128] via WMMA + scatter-add --------
    const int mt  = wv & 3;            // M tile of padded 64 harmonic rows
    const int ntb = (wv >> 2) * 4;     // 4 channel tiles per wave
    const int gi  = mt * 16 + lr;
    for (int e = 0; e < MT; ++e) {
        // A fragment: one ds_load_b128 (wig row, 7 values + zero pad), hi lanes zero
        v8bf al = *(const v8bf*)(wig + (e * 64 + gi) * 8);
        v16bf a;
        #pragma unroll
        for (int i = 0; i < 8; ++i)  a[i] = hi ? f2bf(0.f) : al[i];
        #pragma unroll
        for (int i = 8; i < 16; ++i) a[i] = f2bf(0.f);

        const int tgt = tgtid[e];
        float* outb = out + (size_t)tgt * N_SPH * CCH;
        for (int j = 0; j < 4; ++j) {
            const int col = (ntb + j) * 16 + lr;   // channel
            v8bf bl = *(const v8bf*)(x0t + (e * CCH + col) * 8);
            v16bf b;
            #pragma unroll
            for (int i = 0; i < 8; ++i)  b[i] = hi ? f2bf(0.f) : bl[i];
            #pragma unroll
            for (int i = 8; i < 16; ++i) b[i] = f2bf(0.f);

            v8f acc = {0.f,0.f,0.f,0.f,0.f,0.f,0.f,0.f};
            acc = __builtin_amdgcn_wmma_f32_16x16x32_bf16(false, a, false, b,
                                                          (short)0, acc, false, false);
            #pragma unroll
            for (int r = 0; r < 8; ++r) {
                int g = mt * 16 + r + hi * 8;
                if (g < N_SPH)
                    atomicAdd(outb + g * CCH + col, acc[r] * INV_RESCALE);
            }
        }
    }
}

extern "C" void kernel_launch(void* const* d_in, const int* in_sizes, int n_in,
                              void* d_out, int out_size, void* d_ws, size_t ws_size,
                              hipStream_t stream) {
    const int*   atomic_numbers = (const int*)  d_in[0];
    const float* edge_distance  = (const float*)d_in[1];
    const int*   edge_index     = (const int*)  d_in[2];
    const float* src_table      = (const float*)d_in[3];
    const float* tgt_table      = (const float*)d_in[4];
    const float* w1  = (const float*)d_in[5];
    const float* b1  = (const float*)d_in[6];
    const float* g1  = (const float*)d_in[7];
    const float* be1 = (const float*)d_in[8];
    const float* w2  = (const float*)d_in[9];
    const float* b2  = (const float*)d_in[10];
    const float* g2  = (const float*)d_in[11];
    const float* be2 = (const float*)d_in[12];
    const float* w3  = (const float*)d_in[13];
    const float* b3  = (const float*)d_in[14];
    // d_in[15] = to_m (column map hard-coded from LMAX/MMAX)
    const float* wigner_inv = (const float*)d_in[16];
    float* out = (float*)d_out;

    __bf16* wsb = (__bf16*)d_ws;
    __bf16* w1p = (__bf16*)((char*)d_ws + WS_W1);
    __bf16* w2p = (__bf16*)((char*)d_ws + WS_W2);
    __bf16* w3p = (__bf16*)((char*)d_ws + WS_W3);
    (void)wsb; (void)ws_size; (void)n_in; (void)in_sizes;

    // pre-pack weights to bf16 WMMA-B fragment layout (reused by all 2000 blocks)
    pack_w_kernel<<<(D_IN * HID + 255) / 256, 256, 0, stream>>>(w1, w1p, D_IN, HID, D_IN * HID);
    pack_w_kernel<<<(HID * HID + 255) / 256, 256, 0, stream>>>(w2, w2p, HID, HID, HID * HID);
    pack_w_kernel<<<(HID * W3N + 255) / 256, 256, 0, stream>>>(w3, w3p, HID, W3N, HID * W3N);

    int n4 = out_size / 4;
    zero_out_kernel<<<(n4 + 255) / 256, 256, 0, stream>>>((float4*)out, n4);

    edge_degree_fused_kernel<<<N_EDGES / MT, 256, 0, stream>>>(
        atomic_numbers, edge_distance, edge_index, src_table, tgt_table,
        w1p, w2p, w3p, b1, g1, be1, b2, g2, be2, b3, wigner_inv, out);
}
